// MultiheadAttention_68874095559336
// MI455X (gfx1250) — compile-verified, matrix operand order checked by reading, not run on hardware
//
#include <hip/hip_runtime.h>
#include <hip/hip_bf16.h>

// ---------------------------------------------------------------------------
// MHA forward for MI455X (gfx1250), wave32 + WMMA + async/TDM LDS staging.
//   B=4, N=2048, D=1024, H=16, HD=64
// Pipeline:
//   1) Q = f16(x @ Wq^T + bq)   (B*N, D) f16      [wmma gemm]
//   2) K = f16(x @ Wk^T + bk)   (B*N, D) f16      [wmma gemm]
//   3) Vt = f16(x @ Wv^T + bv) transposed to (B,H,HD,N) f16  [wmma gemm]
//   4) attn: flash-style online softmax; K tile via TENSOR_LOAD_TO_LDS (TDM),
//      V tile via GLOBAL_LOAD_ASYNC_TO_LDS, double-buffered; row-sum via
//      ones-matrix WMMA                                      [wmma]
//   5) out = attn @ Wo^T + bo -> f32 d_out                   [wmma gemm]
// Workspace: 4 buffers of 8192*1024 f16 = 64 MiB total.
// ---------------------------------------------------------------------------

typedef __attribute__((ext_vector_type(16))) _Float16     v16h;
typedef __attribute__((ext_vector_type(8)))  _Float16     v8h;
typedef __attribute__((ext_vector_type(8)))  float        v8f;
typedef __attribute__((ext_vector_type(4)))  int          v4i;
typedef __attribute__((ext_vector_type(8)))  int          v8i;
typedef __attribute__((ext_vector_type(4)))  unsigned int v4u;

constexpr int Bb   = 4;
constexpr int Ntok = 2048;
constexpr int Dm   = 1024;
constexpr int Hh   = 16;
constexpr int HDd  = 64;
constexpr int Mrows = Bb * Ntok;          // 8192

#ifndef __has_builtin
#define __has_builtin(x) 0
#endif

#if __has_builtin(__builtin_amdgcn_global_load_async_to_lds_b128)
#define HAVE_ASYNC_LDS 1
#else
#define HAVE_ASYNC_LDS 0
#endif

#if __has_builtin(__builtin_amdgcn_tensor_load_to_lds)
#define HAVE_TDM 1
#else
#define HAVE_TDM 0
#endif

#define AS1 __attribute__((address_space(1)))
#define AS3 __attribute__((address_space(3)))

// Copy 16 bytes global -> LDS. Async path uses ASYNCcnt-tracked DMA.
__device__ inline void async_copy16(const _Float16* g, _Float16* l) {
#if HAVE_ASYNC_LDS
  __builtin_amdgcn_global_load_async_to_lds_b128((AS1 v4i*)g, (AS3 v4i*)l,
                                                 0, 0);
#else
  *(v8h*)l = *(const v8h*)g;
#endif
}

template <int N>
__device__ inline void wait_async() {
#if HAVE_ASYNC_LDS
#if __has_builtin(__builtin_amdgcn_s_wait_asynccnt)
  __builtin_amdgcn_s_wait_asynccnt(N);
#else
  if constexpr (N == 0)      asm volatile("s_wait_asynccnt 0" ::: "memory");
  else if constexpr (N == 1) asm volatile("s_wait_asynccnt 1" ::: "memory");
  else                       asm volatile("s_wait_asynccnt 2" ::: "memory");
#endif
#endif
}

template <int N>
__device__ inline void wait_tensor() {
#if HAVE_TDM
#if __has_builtin(__builtin_amdgcn_s_wait_tensorcnt)
  __builtin_amdgcn_s_wait_tensorcnt(N);
#else
  if constexpr (N == 0) asm volatile("s_wait_tensorcnt 0" ::: "memory");
  else                  asm volatile("s_wait_tensorcnt 1" ::: "memory");
#endif
#endif
}

#if HAVE_TDM
// One TDM 2D tile load: tile_d1 rows of tile_d0 halves, row stride stride0
// halves, packed contiguously at lds_off.  D# per CDNA5 ISA 8.3/8.4.
// 6-arg builtin (clang-23 prototype): (v4u g0, v8i g1, v4i g2, v4i g3,
//                                      v8i g4, i32 cpol)
__device__ inline void tdm_load_2d(unsigned lds_off, const _Float16* gaddr,
                                   unsigned tensor_d0, unsigned tensor_d1,
                                   unsigned tile_d0, unsigned tile_d1,
                                   unsigned stride0) {
  const unsigned long long ga = (unsigned long long)(size_t)gaddr;
  v4u g0;
  g0[0] = 1u;                                   // count=1, user desc
  g0[1] = lds_off;                              // lds_addr (bytes)
  g0[2] = (unsigned)ga;                         // global_addr[31:0]
  g0[3] = (unsigned)((ga >> 32) & 0x1FFFFFFu)   // global_addr[56:32]
        | (2u << 30);                           // type=2 ("image")
  v8i g1;
  g1[0] = (int)(1u << 16);                      // wg_mask=0, data_size=2B
  g1[1] = (int)((tensor_d0 & 0xFFFFu) << 16);   // tensor_dim0[15:0] @bit48
  g1[2] = (int)((tensor_d0 >> 16) | ((tensor_d1 & 0xFFFFu) << 16));
  g1[3] = (int)((tensor_d1 >> 16) | ((tile_d0 & 0xFFFFu) << 16));
  g1[4] = (int)(tile_d1 & 0xFFFFu);             // tile_dim1, tile_dim2=0
  g1[5] = (int)stride0;                         // tensor_dim0_stride lo32
  g1[6] = 0;
  g1[7] = 0;
  v4i gz4 = {0, 0, 0, 0};                       // groups 2/3 unused (2D)
  v8i gz8 = {0, 0, 0, 0, 0, 0, 0, 0};
  __builtin_amdgcn_tensor_load_to_lds(g0, g1, gz4, gz4, gz8, 0);
}
#endif

// Build a 16x32 f16 A/B fragment (v16h) from two contiguous 8-element chunks.
// Assumed CDNA5 16-bit operand layout (ISA 7.12.2): lanes 0-15 hold row=lane,
// K in {0..7, 16..23}; lanes 16-31 hold row=lane-16, K in {8..15, 24..31}.
__device__ inline v16h make_frag(const _Float16* p0, const _Float16* p1) {
  v8h a = *(const v8h*)p0;
  v8h b = *(const v8h*)p1;
  v16h r;
#pragma unroll
  for (int i = 0; i < 8; ++i) { r[i] = a[i]; r[i + 8] = b[i]; }
  return r;
}

__device__ inline v16h make_frag(const float* p0, const float* p1) {
  v8f a = *(const v8f*)p0;
  v8f b = *(const v8f*)p1;
  v16h r;
#pragma unroll
  for (int i = 0; i < 8; ++i) {
    r[i]     = (_Float16)a[i];
    r[i + 8] = (_Float16)b[i];
  }
  return r;
}

__device__ inline v8f wmma_f16(v16h a, v16h b, v8f c) {
  // (neg_a, A, neg_b, B, c_mod, C, reuse_a, reuse_b)
  return __builtin_amdgcn_wmma_f32_16x16x32_f16(false, a, false, b,
                                                (short)0, c, false, false);
}

// ---------------------------------------------------------------------------
// GEMM + bias:  Y[M=8192, 1024] = A[M,1024] @ W[1024,1024]^T + bias
//   One wave -> one 16x16 output tile; block = 8 waves covering 128 columns.
//   grid = (M/16, 1024/128)
//   TRANS_OUT stores into Vt[(b*H+h)*HD + d][n] instead of Y[m][col].
// ---------------------------------------------------------------------------
template <typename InT, typename OutT, bool TRANS_OUT>
__global__ __launch_bounds__(256) void gemm_bias_kernel(
    const InT* __restrict__ A, const float* __restrict__ W,
    const float* __restrict__ bias, OutT* __restrict__ Y) {
  const int wave = threadIdx.x >> 5;
  const int lane = threadIdx.x & 31;
  const int frow = lane & 15;            // fragment row / column select
  const int fk   = (lane >> 4) << 3;     // 0 or 8: K-chunk select
  const int m0   = blockIdx.x * 16;
  const int n0   = (blockIdx.y * 8 + wave) * 16;

  const InT*   arow = A + (size_t)(m0 + frow) * Dm;
  const float* wrow = W + (size_t)(n0 + frow) * Dm;

  v8f acc = {};
#pragma unroll 4
  for (int k = 0; k < Dm; k += 32) {
    v16h af = make_frag(arow + k + fk, arow + k + fk + 16);
    v16h bf = make_frag(wrow + k + fk, wrow + k + fk + 16);
    acc = wmma_f16(af, bf, acc);
  }

  const int   col = n0 + frow;           // C layout: lane's column is fixed
  const float bc  = bias[col];
#pragma unroll
  for (int r = 0; r < 8; ++r) {
    const int   m = m0 + r + ((lane >> 4) << 3);
    const float v = acc[r] + bc;
    if constexpr (TRANS_OUT) {
      const int bidx = m >> 11;          // / Ntok
      const int n    = m & (Ntok - 1);
      const int h    = col >> 6;         // / HD
      const int d    = col & 63;
      Y[((size_t)((bidx * Hh + h) * HDd + d)) * Ntok + n] = (OutT)v;
    } else {
      Y[(size_t)m * Dm + col] = (OutT)v;
    }
  }
}

// ---------------------------------------------------------------------------
// Fused attention with online softmax.
//   grid.x = B*H, grid.y = N/128; block = 256 (8 waves).
//   One wave: 16 query rows x full head dim (64).
//   K tile staged by TDM (wave 0, TENSORcnt), V tile by per-lane async
//   loads (ASYNCcnt); both double-buffered.  Row sums via ones WMMA.
//   Reference semantics: out = (softmax(QK^T) / sqrt(D)) @ V
// ---------------------------------------------------------------------------
__global__ __launch_bounds__(256) void attn_kernel(
    const _Float16* __restrict__ Q, const _Float16* __restrict__ K,
    const _Float16* __restrict__ Vt, _Float16* __restrict__ O) {
  __shared__ __align__(16) _Float16 kbuf[2][32][64];   // 8 KB
  __shared__ __align__(16) _Float16 vbuf[2][64][32];   // 8 KB
  __shared__ __align__(16) _Float16 pbuf[8][16 * 32];  // 8 KB (per-wave P)

  const int tid  = threadIdx.x;
  const int wave = tid >> 5;
  const int lane = tid & 31;
  const int frow = lane & 15;
  const int fk   = (lane >> 4) << 3;
  const int bh   = blockIdx.x;
  const int b    = bh >> 4;
  const int h    = bh & 15;
  const int q0   = blockIdx.y * 128 + wave * 16;

  const _Float16* Qp = Q  + (size_t)b * Ntok * Dm + (size_t)h * HDd;
  const _Float16* Kp = K  + (size_t)b * Ntok * Dm + (size_t)h * HDd;
  const _Float16* Vp = Vt + (size_t)bh * HDd * Ntok;

  // Stage one 32-key tile: K (32x64) + V (64x32).
  auto issue_tile = [&](int jt, int buf) {
    // V: 256 threads x one 16B chunk -> 1 async op per wave
    const _Float16* gv = Vp + (size_t)(tid >> 2) * Ntok + jt + ((tid & 3) << 3);
    async_copy16(gv, &vbuf[buf][tid >> 2][(tid & 3) << 3]);
#if HAVE_TDM
    // K: single TDM descriptor issued by wave 0
    if (wave == 0) {
      const unsigned koff = (unsigned)(size_t)(AS3 void*)&kbuf[buf][0][0];
      tdm_load_2d(koff, Kp + (size_t)jt * Dm,
                  /*tensor_d0=*/HDd, /*tensor_d1=*/Ntok,
                  /*tile_d0=*/64, /*tile_d1=*/32, /*stride0=*/Dm);
    }
#else
    const _Float16* gk = Kp + (size_t)(jt + (tid >> 3)) * Dm + ((tid & 7) << 3);
    async_copy16(gk, &kbuf[buf][tid >> 3][(tid & 7) << 3]);
#endif
  };
  constexpr int AOPS = HAVE_TDM ? 1 : 2;   // async ops per wave per tile

  // Q fragments for this wave's 16 query rows (d = 0..31 and 32..63)
  const _Float16* qr = Qp + (size_t)(q0 + frow) * Dm;
  const v16h qf0 = make_frag(qr + fk,      qr + fk + 16);
  const v16h qf1 = make_frag(qr + 32 + fk, qr + 32 + fk + 16);

  v16h ones;
#pragma unroll
  for (int i = 0; i < 16; ++i) ones[i] = (_Float16)1.0f;

  float mrow[8], lrow[8];
#pragma unroll
  for (int r = 0; r < 8; ++r) { mrow[r] = -1e30f; lrow[r] = 0.0f; }
  v8f zero = {};
  v8f acc[4] = {zero, zero, zero, zero};   // 16 q-rows x 64 d-cols (f32)

  constexpr int NT = Ntok / 32;            // 64 key tiles
  issue_tile(0, 0);

  for (int it = 0; it < NT; ++it) {
    const int buf = it & 1;
    if (it + 1 < NT) {
      issue_tile((it + 1) * 32, buf ^ 1);  // prefetch next tile
      wait_async<AOPS>();                  // current tile's async ops done
      if (HAVE_TDM && wave == 0) wait_tensor<1>();
    } else {
      wait_async<0>();
      if (HAVE_TDM && wave == 0) wait_tensor<0>();
    }
    __syncthreads();                       // all waves' staging visible

    // ---- S = Q @ K^T for 32 keys (two 16x16 tiles) -----------------------
    const _Float16* kr0 = &kbuf[buf][frow][0];
    const _Float16* kr1 = &kbuf[buf][16 + frow][0];
    v8f s0 = {};
    s0 = wmma_f16(qf0, make_frag(kr0 + fk,      kr0 + fk + 16),      s0);
    s0 = wmma_f16(qf1, make_frag(kr0 + 32 + fk, kr0 + 32 + fk + 16), s0);
    v8f s1 = {};
    s1 = wmma_f16(qf0, make_frag(kr1 + fk,      kr1 + fk + 16),      s1);
    s1 = wmma_f16(qf1, make_frag(kr1 + 32 + fk, kr1 + 32 + fk + 16), s1);

    // ---- online softmax: row max via stage-batched 16-lane butterflies ---
    float tmax[8];
#pragma unroll
    for (int r = 0; r < 8; ++r) tmax[r] = fmaxf(s0[r], s1[r]);
#pragma unroll
    for (int off = 8; off >= 1; off >>= 1) {
      float u[8];
#pragma unroll
      for (int r = 0; r < 8; ++r) u[r] = __shfl_xor(tmax[r], off, 32);
#pragma unroll
      for (int r = 0; r < 8; ++r) tmax[r] = fmaxf(tmax[r], u[r]);
    }
    float sc[8];
#pragma unroll
    for (int r = 0; r < 8; ++r) {
      const float mnew = fmaxf(mrow[r], tmax[r]);
      sc[r]   = __expf(mrow[r] - mnew);
      mrow[r] = mnew;
    }
    // P = exp(S - m) -> f16 into this wave's LDS slab (A-layout staging)
#pragma unroll
    for (int r = 0; r < 8; ++r) {
      const float p0 = __expf(s0[r] - mrow[r]);
      const float p1 = __expf(s1[r] - mrow[r]);
      const int prow = r + ((lane >> 4) << 3);
      pbuf[wave][prow * 32 + frow]      = (_Float16)p0;
      pbuf[wave][prow * 32 + 16 + frow] = (_Float16)p1;
    }
    // rescale accumulators by exp(m_old - m_new), row-wise
#pragma unroll
    for (int t4 = 0; t4 < 4; ++t4)
#pragma unroll
      for (int r = 0; r < 8; ++r) acc[t4][r] *= sc[r];
    __syncthreads();  // order intra-wave LDS store -> cross-lane LDS load

    const v16h pf = make_frag(&pbuf[wave][frow * 32 + fk],
                              &pbuf[wave][frow * 32 + fk + 16]);

    // ---- row sums on the matrix pipe: rs = P @ ones ----------------------
    v8f rs = wmma_f16(pf, ones, zero);
#pragma unroll
    for (int r = 0; r < 8; ++r) lrow[r] = lrow[r] * sc[r] + rs[r];

    // ---- acc += P(16x32) @ V(32x64) --------------------------------------
#pragma unroll
    for (int t4 = 0; t4 < 4; ++t4) {
      const _Float16* vr = &vbuf[buf][t4 * 16 + frow][0];
      acc[t4] = wmma_f16(pf, make_frag(vr + fk, vr + fk + 16), acc[t4]);
    }
    __syncthreads();  // all waves done with buf before it is re-staged
  }

  // ---- normalize (1/rowsum) and apply reference's 1/sqrt(D)=1/32 ---------
#pragma unroll
  for (int r = 0; r < 8; ++r) {
    const float inv  = 1.0f / (lrow[r] * 32.0f);
    const int   qrow = q0 + r + ((lane >> 4) << 3);
#pragma unroll
    for (int t4 = 0; t4 < 4; ++t4) {
      const int col = h * HDd + t4 * 16 + frow;
      O[(size_t)(b * Ntok + qrow) * Dm + col] = (_Float16)(acc[t4][r] * inv);
    }
  }
}

// ---------------------------------------------------------------------------
extern "C" void kernel_launch(void* const* d_in, const int* in_sizes, int n_in,
                              void* d_out, int out_size, void* d_ws,
                              size_t ws_size, hipStream_t stream) {
  (void)in_sizes; (void)n_in; (void)out_size; (void)ws_size;
  const float* x  = (const float*)d_in[0];
  const float* Wq = (const float*)d_in[1];
  const float* bq = (const float*)d_in[2];
  const float* Wk = (const float*)d_in[3];
  const float* bk = (const float*)d_in[4];
  const float* Wv = (const float*)d_in[5];
  const float* bv = (const float*)d_in[6];
  const float* Wo = (const float*)d_in[7];
  const float* bo = (const float*)d_in[8];
  float* out = (float*)d_out;

  // Workspace carve-up: 4 x (8192*1024) f16 = 64 MiB
  const size_t elems = (size_t)Mrows * Dm;
  _Float16* Qb = (_Float16*)d_ws;
  _Float16* Kb = Qb + elems;
  _Float16* Vt = Kb + elems;
  _Float16* Ob = Vt + elems;

  dim3 blk(256);
  dim3 gg(Mrows / 16, Dm / 128);  // (512, 8)

  gemm_bias_kernel<float, _Float16, false><<<gg, blk, 0, stream>>>(x, Wq, bq, Qb);
  gemm_bias_kernel<float, _Float16, false><<<gg, blk, 0, stream>>>(x, Wk, bk, Kb);
  gemm_bias_kernel<float, _Float16, true ><<<gg, blk, 0, stream>>>(x, Wv, bv, Vt);

  attn_kernel<<<dim3(Bb * Hh, Ntok / 128), blk, 0, stream>>>(Qb, Kb, Vt, Ob);

  gemm_bias_kernel<_Float16, float, false><<<gg, blk, 0, stream>>>(Ob, Wo, bo, out);
}
